// GAP_net_49409303773203
// MI455X (gfx1250) — matrix operand order repaired and checked by reading, not compiled, not run on hardware
//
#include <hip/hip_runtime.h>
#include <hip/hip_bf16.h>
#include <math.h>

typedef __attribute__((ext_vector_type(16))) _Float16 v16h;
typedef __attribute__((ext_vector_type(8)))  _Float16 v8h;
typedef __attribute__((ext_vector_type(8)))  float    v8f;

#define BATCH 4
#define CH    28
#define IMG_H 256
#define IMG_W 310
#define STEP  2
#define NSTAGE 9

// UNet spatial geometry (W padded 310 -> 312, /8 friendly)
#define HP0 256
#define WP0 312
#define HP1 128
#define WP1 156
#define HP2 64
#define WP2 78

// ---------------------------------------------------------------------------
// x = y * Phi   (initial estimate)
// ---------------------------------------------------------------------------
__global__ void k_init_x(const float* __restrict__ y, const float* __restrict__ Phi,
                         float* __restrict__ x) {
    int idx = blockIdx.x * blockDim.x + threadIdx.x;
    int total = BATCH * CH * IMG_H * IMG_W;
    if (idx >= total) return;
    int hw = idx % (IMG_H * IMG_W);
    int b  = (idx / (IMG_H * IMG_W)) / CH;
    x[idx] = y[(size_t)b * IMG_H * IMG_W + hw] * Phi[idx];
}

// ---------------------------------------------------------------------------
// GAP projection: x += ((y - sum_c x*Phi)/Phi_s) * Phi     (in place, per column)
// ---------------------------------------------------------------------------
__global__ void k_project(const float* __restrict__ y, const float* __restrict__ Phi,
                          const float* __restrict__ Phis, float* __restrict__ x) {
    int idx = blockIdx.x * blockDim.x + threadIdx.x;
    int total = BATCH * IMG_H * IMG_W;
    if (idx >= total) return;
    int b  = idx / (IMG_H * IMG_W);
    int hw = idx % (IMG_H * IMG_W);
    size_t base = (size_t)b * CH * IMG_H * IMG_W + hw;
    float yb = 0.f;
    for (int c = 0; c < CH; ++c) {
        size_t o = base + (size_t)c * IMG_H * IMG_W;
        yb += x[o] * Phi[o];
    }
    float r = (y[idx] - yb) / Phis[idx];
    for (int c = 0; c < CH; ++c) {
        size_t o = base + (size_t)c * IMG_H * IMG_W;
        x[o] += r * Phi[o];
    }
}

// ---------------------------------------------------------------------------
// Build UNet input: shift_back_3d + reflect-pad W(310->312) + f32->f16,
// channels-last [B][258][314][32] with zeroed halo & zero channels 28..31.
// ---------------------------------------------------------------------------
__global__ void k_build_in0(const float* __restrict__ x, _Float16* __restrict__ in0) {
    int idx = blockIdx.x * blockDim.x + threadIdx.x;
    int total = BATCH * (HP0 + 2) * (WP0 + 2);
    if (idx >= total) return;
    int xs   = idx % (WP0 + 2);
    int rest = idx / (WP0 + 2);
    int ys   = rest % (HP0 + 2);
    int b    = rest / (HP0 + 2);
    _Float16 out[32] __attribute__((aligned(16)));
    if (ys == 0 || ys == HP0 + 1 || xs == 0 || xs == WP0 + 1) {
        #pragma unroll
        for (int c = 0; c < 32; ++c) out[c] = (_Float16)0.f;
    } else {
        int yy = ys - 1;
        int wp = xs - 1;                                   // 0..311
        int wsrc = (wp < IMG_W) ? wp : (wp == IMG_W ? IMG_W - 2 : IMG_W - 3); // reflect
        const float* xb = x + (size_t)b * CH * IMG_H * IMG_W + (size_t)yy * IMG_W;
        for (int c = 0; c < CH; ++c) {
            int v = (wsrc + STEP * c) % IMG_W;             // shift_back
            out[c] = (_Float16)xb[(size_t)c * IMG_H * IMG_W + v];
        }
        #pragma unroll
        for (int c = CH; c < 32; ++c) out[c] = (_Float16)0.f;
    }
    _Float16* dst = in0 + (size_t)idx * 32;
    #pragma unroll
    for (int k = 0; k < 4; ++k) *(v8h*)(dst + k * 8) = *(const v8h*)(out + k * 8);
}

// ---------------------------------------------------------------------------
// Weight repack: f32 [O][I][3][3] -> f16 B-fragments [t][cb][nb][lane][16],
// element e of lane: K = cb*32 + 16*(lane>>4) + e, N = nb*16 + (lane&15).
// ---------------------------------------------------------------------------
__global__ void k_pack3x3(const float* __restrict__ w, _Float16* __restrict__ dst,
                          int I, int O, int numCb, int numNb) {
    int idx = blockIdx.x * blockDim.x + threadIdx.x;
    int total = 9 * numCb * numNb * 512;
    if (idx >= total) return;
    int e    = idx & 15;
    int lane = (idx >> 4) & 31;
    int rest = idx >> 9;
    int nb = rest % numNb; rest /= numNb;
    int cb = rest % numCb;
    int t  = rest / numCb;
    int ci = cb * 32 + (lane >> 4) * 16 + e;
    int co = nb * 16 + (lane & 15);
    float v = (ci < I && co < O) ? w[((size_t)co * I + ci) * 9 + t] : 0.f;
    dst[idx] = (_Float16)v;
}

// ConvTranspose weights: f32 [I][O][2][2] -> same fragment layout, 4 taps.
__global__ void k_packT(const float* __restrict__ w, _Float16* __restrict__ dst,
                        int I, int O, int numCb, int numNb) {
    int idx = blockIdx.x * blockDim.x + threadIdx.x;
    int total = 4 * numCb * numNb * 512;
    if (idx >= total) return;
    int e    = idx & 15;
    int lane = (idx >> 4) & 31;
    int rest = idx >> 9;
    int nb = rest % numNb; rest /= numNb;
    int cb = rest % numCb;
    int t  = rest / numCb;
    int ci = cb * 32 + (lane >> 4) * 16 + e;
    int co = nb * 16 + (lane & 15);
    float v = (ci < I && co < O) ? w[((size_t)ci * O + co) * 4 + t] : 0.f;
    dst[idx] = (_Float16)v;
}

__device__ inline v16h cat8(v8h a, v8h b) {
    v16h r;
    #pragma unroll
    for (int i = 0; i < 8; ++i) { r[i] = a[i]; r[i + 8] = b[i]; }
    return r;
}

#define WMMA_F16(A, B, C) \
    __builtin_amdgcn_wmma_f32_16x16x32_f16(false, (A), false, (B), (short)0, (C), false, false)

// ---------------------------------------------------------------------------
// 3x3 SAME conv as 9 per-tap WMMA GEMMs. Channels-last f16 activations with
// 1-pixel zero halo. Block = one 32-pixel row tile; each wave owns 16 output
// channels and runs TWO independent accumulator chains (two 16-pixel M-tiles
// sharing each B fragment) to cover the WMMA D->src hazard slots.
// B-fragment pointer is a single induction variable (constant stride) and is
// software-pipelined one step ahead of the WMMAs.
// ---------------------------------------------------------------------------
__global__ void k_conv3x3(const _Float16* __restrict__ src0, int c0,
                          const _Float16* __restrict__ src1, int c1,
                          const _Float16* __restrict__ wfrag,
                          const float* __restrict__ bias,
                          _Float16* __restrict__ out,
                          int Hp, int Wp, int Cout, int relu) {
    extern __shared__ _Float16 smem[];
    const int Cin  = c0 + c1;
    const int wave = threadIdx.x >> 5;
    const int lane = threadIdx.x & 31;
    const int tilesPerRow = (Wp + 31) >> 5;
    const int y  = blockIdx.x / tilesPerRow;
    const int x0 = (blockIdx.x % tilesPerRow) << 5;
    const int b  = blockIdx.z;
    const size_t WS = (size_t)(Wp + 2);
    const _Float16* s0 = src0 + (size_t)b * (Hp + 2) * WS * c0;
    const _Float16* s1 = src1 ? (src1 + (size_t)b * (Hp + 2) * WS * c1) : (const _Float16*)0;

    // ---- cooperative LDS stage: [3][34][Cin]
    const int cin8 = Cin >> 3;
    const int nchunk = 102 * cin8;
    for (int i = threadIdx.x; i < nchunk; i += blockDim.x) {
        int cb8 = i % cin8;
        int pix = i / cin8;
        int dy = pix / 34, dx = pix - dy * 34;
        int gx = x0 + dx; if (gx > Wp + 1) gx = Wp + 1;   // clamp last partial tile
        int ci = cb8 << 3;
        v8h v;
        if (ci < c0) v = *(const v8h*)(s0 + ((size_t)(y + dy) * WS + gx) * c0 + ci);
        else         v = *(const v8h*)(s1 + ((size_t)(y + dy) * WS + gx) * c1 + (ci - c0));
        *(v8h*)(smem + (size_t)pix * Cin + ci) = v;
    }
    __syncthreads();

    const int hi = lane >> 4;
    const int ln = lane & 15;
    const int numCb = Cin >> 5;
    const int numNb = Cout >> 4;
    const int nb = wave;
    const int bstride = numNb * 512;          // halves between consecutive (t,cb) frags

    const _Float16* bp = wfrag + ((size_t)nb * 32 + lane) * 16;
    v16h Bcur = *(const v16h*)bp;
    bp += bstride;

    v8f acc0 = {0.f, 0.f, 0.f, 0.f, 0.f, 0.f, 0.f, 0.f};
    v8f acc1 = acc0;
    for (int t = 0; t < 9; ++t) {
        int dy = t / 3, dx = t - dy * 3;
        const _Float16* arow = smem + ((size_t)(dy * 34) + (ln + dx)) * Cin + hi * 16;
        for (int cb = 0; cb < numCb; ++cb) {
            const _Float16* ap = arow + cb * 32;
            v16h A0 = cat8(*(const v8h*)ap,                  *(const v8h*)(ap + 8));
            v16h A1 = cat8(*(const v8h*)(ap + 16 * Cin),     *(const v8h*)(ap + 16 * Cin + 8));
            v16h Bn = *(const v16h*)bp;      // pipelined next fragment (tail-padded)
            bp += bstride;
            acc0 = WMMA_F16(A0, Bcur, acc0);
            acc1 = WMMA_F16(A1, Bcur, acc1);
            Bcur = Bn;
        }
    }

    const int co = nb * 16 + ln;
    const float bv = bias[co];
    _Float16* ob = out + (size_t)b * (Hp + 2) * WS * Cout + ((size_t)(y + 1) * WS + 1) * Cout + co;
    #pragma unroll
    for (int r = 0; r < 8; ++r) {
        int m = r + hi * 8;                  // D layout: M = r + 8*(lane>>4)
        int gx0 = x0 + m;
        int gx1 = x0 + 16 + m;
        if (gx0 < Wp) {
            float v = acc0[r] + bv;
            if (relu) v = fmaxf(v, 0.f);
            ob[(size_t)gx0 * Cout] = (_Float16)v;
        }
        if (gx1 < Wp) {
            float v = acc1[r] + bv;
            if (relu) v = fmaxf(v, 0.f);
            ob[(size_t)gx1 * Cout] = (_Float16)v;
        }
    }
}

// ---------------------------------------------------------------------------
// ConvTranspose2d k=2 s=2 (non-overlapping): 4 per-tap 1x1 WMMA GEMMs.
// 32 input pixels per wave-tile -> 8 independent accumulator chains.
// ---------------------------------------------------------------------------
__global__ void k_convT2x2(const _Float16* __restrict__ src, int Cin,
                           const _Float16* __restrict__ wfrag,
                           const float* __restrict__ bias,
                           _Float16* __restrict__ out,
                           int h, int w, int Cout) {
    const int wave = threadIdx.x >> 5;
    const int lane = threadIdx.x & 31;
    const int tilesPerRow = (w + 31) >> 5;
    const int y  = blockIdx.x / tilesPerRow;
    const int x0 = (blockIdx.x % tilesPerRow) << 5;
    const int b  = blockIdx.z;
    const size_t WSi = (size_t)(w + 2);
    const size_t WSo = (size_t)(2 * w + 2);
    const _Float16* sb = src + (size_t)b * (h + 2) * WSi * Cin;
    const int hi = lane >> 4, ln = lane & 15;
    const int numCb = Cin >> 5;
    const int numNb = Cout >> 4;
    const int nb = wave;
    const int cstride = numNb * 512;          // halves between cb steps (fixed t)
    const int tstride = numCb * numNb * 512;  // halves between taps

    int gxi0 = x0 + ln;      if (gxi0 >= w) gxi0 = w - 1;   // clamp (stores masked)
    int gxi1 = x0 + 16 + ln; if (gxi1 >= w) gxi1 = w - 1;
    const _Float16* ap0 = sb + ((size_t)(y + 1) * WSi + (gxi0 + 1)) * Cin + hi * 16;
    const _Float16* ap1 = sb + ((size_t)(y + 1) * WSi + (gxi1 + 1)) * Cin + hi * 16;

    const _Float16* bp = wfrag + ((size_t)nb * 32 + lane) * 16;
    v16h Bc[4];
    #pragma unroll
    for (int t = 0; t < 4; ++t) Bc[t] = *(const v16h*)(bp + (size_t)t * tstride);

    v8f zero = {0.f, 0.f, 0.f, 0.f, 0.f, 0.f, 0.f, 0.f};
    v8f acc[8];
    #pragma unroll
    for (int t = 0; t < 8; ++t) acc[t] = zero;

    for (int cb = 0; cb < numCb; ++cb) {
        v16h A0 = cat8(*(const v8h*)(ap0 + cb * 32), *(const v8h*)(ap0 + cb * 32 + 8));
        v16h A1 = cat8(*(const v8h*)(ap1 + cb * 32), *(const v8h*)(ap1 + cb * 32 + 8));
        v16h Bn[4];
        #pragma unroll
        for (int t = 0; t < 4; ++t)        // pipelined next cb (tail-padded)
            Bn[t] = *(const v16h*)(bp + (size_t)t * tstride + (size_t)(cb + 1) * cstride);
        #pragma unroll
        for (int t = 0; t < 4; ++t) {
            acc[2 * t]     = WMMA_F16(A0, Bc[t], acc[2 * t]);
            acc[2 * t + 1] = WMMA_F16(A1, Bc[t], acc[2 * t + 1]);
        }
        #pragma unroll
        for (int t = 0; t < 4; ++t) Bc[t] = Bn[t];
    }

    const int co = nb * 16 + ln;
    const float bv = bias[co];
    _Float16* ob = out + (size_t)b * (2 * h + 2) * WSo * Cout;
    #pragma unroll
    for (int t = 0; t < 4; ++t) {
        int i = t >> 1, j = t & 1;
        #pragma unroll
        for (int r = 0; r < 8; ++r) {
            int m = r + hi * 8;
            int gx0 = x0 + m;
            int gx1 = x0 + 16 + m;
            if (gx0 < w) {
                float v = fmaxf(acc[2 * t][r] + bv, 0.f);
                ob[((size_t)(2 * y + i + 1) * WSo + (2 * gx0 + j + 1)) * Cout + co] = (_Float16)v;
            }
            if (gx1 < w) {
                float v = fmaxf(acc[2 * t + 1][r] + bv, 0.f);
                ob[((size_t)(2 * y + i + 1) * WSo + (2 * gx1 + j + 1)) * Cout + co] = (_Float16)v;
            }
        }
    }
}

// ---------------------------------------------------------------------------
// 2x2 maxpool (stride 2), halo buffers in & out.
// ---------------------------------------------------------------------------
__global__ void k_pool(const _Float16* __restrict__ src, _Float16* __restrict__ dst,
                       int h, int w, int C) {
    int h2 = h >> 1, w2 = w >> 1;
    int idx = blockIdx.x * blockDim.x + threadIdx.x;
    int total = BATCH * h2 * w2 * C;
    if (idx >= total) return;
    int c = idx % C; int rest = idx / C;
    int x2 = rest % w2; rest /= w2;
    int y2 = rest % h2;
    int b  = rest / h2;
    const _Float16* s = src + (size_t)b * (h + 2) * (w + 2) * C;
    size_t WS = (size_t)(w + 2);
    float m = -INFINITY;
    #pragma unroll
    for (int i = 0; i < 2; ++i)
        #pragma unroll
        for (int j = 0; j < 2; ++j) {
            float v = (float)s[((size_t)(2 * y2 + i + 1) * WS + (2 * x2 + j + 1)) * C + c];
            m = fmaxf(m, v);
        }
    dst[((size_t)b * (h2 + 2) * (w2 + 2) + (size_t)(y2 + 1) * (w2 + 2) + (x2 + 1)) * C + c]
        = (_Float16)m;
}

// ---------------------------------------------------------------------------
// Fused stage epilogue: 1x1 conv(32->28) + tanh + residual, with shift_3d
// folded in:  x_next[c,y,w] = x[c,y,w] + tanh(conv)[c,y,(w-2c) mod 310].
// Final stage: out[c,y,w<256] = x[c,y,(w+2c) mod 310] + tanh(conv)[c,y,w].
// ---------------------------------------------------------------------------
__global__ void k_last(const _Float16* __restrict__ up1,
                       const float* __restrict__ wl, const float* __restrict__ bl,
                       float* __restrict__ x, float* __restrict__ outp, int fin) {
    int Wl = fin ? 256 : IMG_W;
    int idx = blockIdx.x * blockDim.x + threadIdx.x;
    int total = BATCH * CH * IMG_H * Wl;
    if (idx >= total) return;
    int wc = idx % Wl; int rest = idx / Wl;
    int yy = rest % IMG_H; rest /= IMG_H;
    int c  = rest % CH;
    int b  = rest / CH;
    int u = fin ? wc : (((wc - STEP * c) % IMG_W) + IMG_W) % IMG_W;
    const _Float16* p = up1 + (((size_t)b * (HP0 + 2) + (yy + 1)) * (WP0 + 2) + (u + 1)) * 32;
    float a = bl[c];
    const float* wr = wl + c * 32;
    #pragma unroll
    for (int k = 0; k < 32; ++k) a += (float)p[k] * wr[k];
    float t = tanhf(a);
    size_t xoff = (((size_t)b * CH + c) * IMG_H + yy) * IMG_W;
    if (fin) {
        int v = (wc + STEP * c) % IMG_W;
        outp[(((size_t)b * CH + c) * IMG_H + yy) * 256 + wc] = x[xoff + v] + t;
    } else {
        x[xoff + wc] += t;
    }
}

// ---------------------------------------------------------------------------
extern "C" void kernel_launch(void* const* d_in, const int* in_sizes, int n_in,
                              void* d_out, int out_size, void* d_ws, size_t ws_size,
                              hipStream_t stream) {
    (void)n_in; (void)out_size; (void)ws_size;
    const float* y    = (const float*)d_in[0];
    const float* Phi  = (const float*)d_in[1];
    const float* Phis = (const float*)d_in[2];

    // Param flattening order detection: insertion (d1,d2,d3,u2,u1,up2,up1,last)
    // vs alphabetical (d1,d2,d3,last,u1,u2,up1,up2). in_sizes[3+12]: u2.w=32768 vs last.w=896.
    bool sorted = (in_sizes[3 + 12] == 28 * 32);
    // group ids: 0=d1 1=d2 2=d3 3=u2 4=u1 5=up2 6=up1 7=last
    const int ofs_ins[8] = {0, 4, 8, 12, 14, 16, 20, 24};
    const int ofs_srt[8] = {0, 4, 8, 16, 14, 22, 18, 12};
    const int* G = sorted ? ofs_srt : ofs_ins;
    auto P = [&](int s, int grp, int k) -> const float* {
        return (const float*)d_in[3 + s * 26 + G[grp] + k];
    };

    // ---- workspace layout
    char* ws = (char*)d_ws;
    size_t off = 0;
    auto alloc = [&](size_t bytes) -> void* {
        void* p = ws + off; off += (bytes + 255) & ~(size_t)255; return p;
    };
    float* X = (float*)alloc((size_t)BATCH * CH * IMG_H * IMG_W * 4);
    size_t zstart = off;
    const size_t full32 = (size_t)BATCH * (HP0 + 2) * (WP0 + 2) * 32 * 2;
    const size_t half32 = (size_t)BATCH * (HP1 + 2) * (WP1 + 2) * 32 * 2;
    const size_t half64 = (size_t)BATCH * (HP1 + 2) * (WP1 + 2) * 64 * 2;
    const size_t qu64   = (size_t)BATCH * (HP2 + 2) * (WP2 + 2) * 64 * 2;
    const size_t qu128  = (size_t)BATCH * (HP2 + 2) * (WP2 + 2) * 128 * 2;
    _Float16* F0 = (_Float16*)alloc(full32);   // in0 -> u1o -> up1
    _Float16* F1 = (_Float16*)alloc(full32);   // c1a -> up1a
    _Float16* F2 = (_Float16*)alloc(full32);   // c1 (skip)
    _Float16* P1 = (_Float16*)alloc(half32);   // pool(c1)
    _Float16* B1 = (_Float16*)alloc(half64);   // c2a -> u2o -> up2
    _Float16* B2 = (_Float16*)alloc(half64);   // c2 (skip)
    _Float16* B3 = (_Float16*)alloc(half64);   // up2a
    _Float16* Q1 = (_Float16*)alloc(qu64);     // pool(c2)
    _Float16* Q2 = (_Float16*)alloc(qu128);    // c3a
    _Float16* Q3 = (_Float16*)alloc(qu128);    // c3
    size_t zend = off;

    // fragment arrays: id 0..11
    const int fI[12] = {28, 32, 32, 64, 64, 128, 128, 64, 128, 64, 64, 32};
    const int fO[12] = {32, 32, 64, 64, 128, 128, 64, 32, 64, 64, 32, 32};
    const int fT[12] = {9, 9, 9, 9, 9, 9, 4, 4, 9, 9, 9, 9};
    _Float16* frag[NSTAGE][12];
    for (int s = 0; s < NSTAGE; ++s)
        for (int id = 0; id < 12; ++id) {
            size_t halves = (size_t)fT[id] * ((fI[id] + 31) / 32) * (fO[id] / 16) * 512;
            frag[s][id] = (_Float16*)alloc(halves * 2);
        }
    alloc(16384);   // tail pad: B-fragment pipeline over-reads <= one fragment

    // zero activation halos (and padding channels stay clean across stages)
    hipMemsetAsync(ws + zstart, 0, zend - zstart, stream);

    // ---- pack all stage weights into WMMA B-fragment layout
    for (int s = 0; s < NSTAGE; ++s) {
        const float* wsrc[12] = {
            P(s, 0, 0), P(s, 0, 2),   // d1 conv1/conv2
            P(s, 1, 0), P(s, 1, 2),   // d2
            P(s, 2, 0), P(s, 2, 2),   // d3
            P(s, 3, 0),               // u2 (transpose)
            P(s, 4, 0),               // u1 (transpose)
            P(s, 5, 0), P(s, 5, 2),   // up2
            P(s, 6, 0), P(s, 6, 2)    // up1
        };
        for (int id = 0; id < 12; ++id) {
            int numCb = (fI[id] + 31) / 32, numNb = fO[id] / 16;
            int total = fT[id] * numCb * numNb * 512;
            int blocks = (total + 255) / 256;
            if (fT[id] == 9)
                k_pack3x3<<<blocks, 256, 0, stream>>>(wsrc[id], frag[s][id], fI[id], fO[id], numCb, numNb);
            else
                k_packT<<<blocks, 256, 0, stream>>>(wsrc[id], frag[s][id], fI[id], fO[id], numCb, numNb);
        }
    }

    auto conv = [&](const _Float16* a, int ca, const _Float16* b2, int cb2, int s, int id,
                    const float* bias, _Float16* o, int Hp, int Wp, int Cout) {
        int tiles = Hp * ((Wp + 31) / 32);
        dim3 g(tiles, 1, BATCH), blk(32 * (Cout / 16));
        size_t sh = (size_t)102 * (ca + cb2) * 2;
        k_conv3x3<<<g, blk, sh, stream>>>(a, ca, b2, cb2, frag[s][id], bias, o, Hp, Wp, Cout, 1);
    };

    // ---- init
    {
        int total = BATCH * CH * IMG_H * IMG_W;
        k_init_x<<<(total + 255) / 256, 256, 0, stream>>>(y, Phi, X);
    }
    int nPix = BATCH * IMG_H * IMG_W;
    int nIn0 = BATCH * (HP0 + 2) * (WP0 + 2);

    for (int s = 0; s < NSTAGE; ++s) {
        k_project<<<(nPix + 255) / 256, 256, 0, stream>>>(y, Phi, Phis, X);
        k_build_in0<<<(nIn0 + 255) / 256, 256, 0, stream>>>(X, F0);

        conv(F0, 32, nullptr, 0, s, 0, P(s, 0, 1), F1, HP0, WP0, 32);     // d1 conv1
        conv(F1, 32, nullptr, 0, s, 1, P(s, 0, 3), F2, HP0, WP0, 32);     // d1 conv2 -> c1
        { int t = BATCH * HP1 * WP1 * 32;
          k_pool<<<(t + 255) / 256, 256, 0, stream>>>(F2, P1, HP0, WP0, 32); }
        conv(P1, 32, nullptr, 0, s, 2, P(s, 1, 1), B1, HP1, WP1, 64);     // d2 conv1
        conv(B1, 64, nullptr, 0, s, 3, P(s, 1, 3), B2, HP1, WP1, 64);     // d2 conv2 -> c2
        { int t = BATCH * HP2 * WP2 * 64;
          k_pool<<<(t + 255) / 256, 256, 0, stream>>>(B2, Q1, HP1, WP1, 64); }
        conv(Q1, 64,  nullptr, 0, s, 4, P(s, 2, 1), Q2, HP2, WP2, 128);   // d3 conv1
        conv(Q2, 128, nullptr, 0, s, 5, P(s, 2, 3), Q3, HP2, WP2, 128);   // d3 conv2 -> c3
        { int tiles = HP2 * ((WP2 + 31) / 32);
          k_convT2x2<<<dim3(tiles, 1, BATCH), 32 * 4, 0, stream>>>(Q3, 128, frag[s][6],
                                                                   P(s, 3, 1), B1, HP2, WP2, 64); }
        conv(B1, 64, B2, 64, s, 8, P(s, 5, 1), B3, HP1, WP1, 64);         // up2 conv1 (concat)
        conv(B3, 64, nullptr, 0, s, 9, P(s, 5, 3), B1, HP1, WP1, 64);     // up2 conv2
        { int tiles = HP1 * ((WP1 + 31) / 32);
          k_convT2x2<<<dim3(tiles, 1, BATCH), 32 * 2, 0, stream>>>(B1, 64, frag[s][7],
                                                                   P(s, 4, 1), F0, HP1, WP1, 32); }
        conv(F0, 32, F2, 32, s, 10, P(s, 6, 1), F1, HP0, WP0, 32);        // up1 conv1 (concat)
        conv(F1, 32, nullptr, 0, s, 11, P(s, 6, 3), F0, HP0, WP0, 32);    // up1 conv2

        int fin = (s == NSTAGE - 1);
        int Wl = fin ? 256 : IMG_W;
        int nl = BATCH * CH * IMG_H * Wl;
        k_last<<<(nl + 255) / 256, 256, 0, stream>>>(F0, P(s, 7, 0), P(s, 7, 1),
                                                     X, (float*)d_out, fin);
    }
}
